// TwoHeadAttention_34703335751761
// MI455X (gfx1250) — matrix operand order repaired and checked by reading, not compiled, not run on hardware
//
#include <hip/hip_runtime.h>

// ---------------- problem constants ----------------
constexpr int D  = 1024;   // d_model
constexpr int NH = 16;     // heads
constexpr int DH = 64;     // head dim
constexpr int BS = 2;      // batch
constexpr int S  = 2048;   // seq
constexpr int M  = BS * S; // 4096 rows in all big GEMMs

typedef unsigned short u16;
typedef __attribute__((ext_vector_type(16))) __bf16 v16bf;
typedef __attribute__((ext_vector_type(8)))  __bf16 v8bf;
typedef __attribute__((ext_vector_type(8)))  float  v8f;

// ---- gfx1250 async global->LDS copy (guarded; falls back to sync staging) ----
#if defined(__has_builtin)
#  if __has_builtin(__builtin_amdgcn_global_load_async_to_lds_b128) && \
      __has_builtin(__builtin_amdgcn_s_wait_asynccnt)
#    define USE_ASYNC 1
#  endif
#endif

typedef int v4i_t __attribute__((vector_size(16)));
typedef __attribute__((address_space(1))) v4i_t* gp128;   // global int4*
typedef __attribute__((address_space(3))) v4i_t* lp128;   // LDS int4*

__device__ __forceinline__ void cp16(u16* ldst, const u16* gsrc) {
#ifdef USE_ASYNC
  __builtin_amdgcn_global_load_async_to_lds_b128((gp128)gsrc, (lp128)ldst, 0, 0);
#else
  *(uint4*)ldst = *(const uint4*)gsrc;
#endif
}

__device__ __forceinline__ void wait_async() {
#ifdef USE_ASYNC
  __builtin_amdgcn_s_wait_asynccnt(0);
#endif
}

// float -> bf16 round-to-nearest-even (manual, avoids __bf16 arithmetic)
__device__ __forceinline__ u16 f2bf(float f) {
  union { float f; unsigned u; } c; c.f = f;
  unsigned u = c.u;
  return (u16)((u + 0x7FFFu + ((u >> 16) & 1u)) >> 16);
}

// assemble a 16-element bf16 fragment from two aligned 8-element chunks
__device__ __forceinline__ v16bf frag16(const u16* p0, const u16* p1) {
  v8bf lo = *(const v8bf*)p0;
  v8bf hi = *(const v8bf*)p1;
  return __builtin_shufflevector(lo, hi, 0,1,2,3,4,5,6,7,8,9,10,11,12,13,14,15);
}

__device__ __forceinline__ v8f wmma_bf16(v16bf a, v16bf b, v8f c) {
  // D = A(16x32 bf16) * B(32x16 bf16) + C(16x16 f32)
  return __builtin_amdgcn_wmma_f32_16x16x32_bf16(
      /*neg_a=*/false, a, /*neg_b=*/false, b,
      /*c_mod=*/(short)0, c, /*reuse_a=*/false, /*reuse_b=*/false);
}

// ---------------- small helper kernels ----------------
__global__ void k_cvt(const float* __restrict__ in, u16* __restrict__ out, int n) {
  int i = blockIdx.x * 256 + threadIdx.x;
  if (i < n) out[i] = f2bf(in[i]);
}

// out[n*D + k] = bf16(in[k*D + n])  (basis transpose, done once)
__global__ void k_transpose(const float* __restrict__ in, u16* __restrict__ out) {
  int i = blockIdx.x * 256 + threadIdx.x;   // over D*D
  int nr = i >> 10;          // /D
  int kk = i & (D - 1);      // %D
  out[i] = f2bf(in[(size_t)kk * D + nr]);
}

// out[i] = bf16(y[i] * scale[i % D])
__global__ void k_scale(const float* __restrict__ y, const float* __restrict__ scale,
                        u16* __restrict__ out, int n) {
  int i = blockIdx.x * 256 + threadIdx.x;
  if (i < n) out[i] = f2bf(y[i] * scale[i & (D - 1)]);
}

// ---------------- bf16 WMMA GEMM: C = A(MxK) * W,  Wt = W^T stored [N][K] ----------------
// 128x128 C tile per block, 256 threads (8 waves), each wave 64x32 of C.
// Double-buffered LDS tiles filled by async global->LDS copies (overlaps WMMA).
constexpr int LDT = 40;  // padded LDS row stride (elements); 80B keeps 16B alignment

template <int EPI>  // 0: store f32, 1: store bf16
__global__ __launch_bounds__(256) void k_gemm_bf16(
    const u16* __restrict__ A,   // [M][D] bf16
    const u16* __restrict__ Wt,  // [D][D] bf16 (W^T row-major)
    float* __restrict__ outF, u16* __restrict__ outB) {
  __shared__ __align__(16) u16 sA[2][128 * LDT];
  __shared__ __align__(16) u16 sB[2][128 * LDT];

  const int tid  = threadIdx.x;
  const int lane = tid & 31;
  const int wid  = tid >> 5;    // 0..7
  const int wr   = wid >> 2;    // 0..1 (row group of 64)
  const int wc   = wid & 3;     // 0..3 (col group of 32)
  const int m0   = blockIdx.y * 128;
  const int n0   = blockIdx.x * 128;

  v8f acc[4][2];
#pragma unroll
  for (int i = 0; i < 4; ++i)
#pragma unroll
    for (int j = 0; j < 2; ++j) acc[i][j] = (v8f){0,0,0,0,0,0,0,0};

  const int lrow = tid >> 1;          // 0..127
  const int lcol = (tid & 1) * 16;    // 0 or 16

  const int mlane = lane & 15;
  const int hsel  = lane >> 4;        // lane half
  const int khA   = hsel * 8;         // A-fragment K sub-chunk
  const int kbB   = hsel * 16;        // unused; B fragments use khA like A

  const u16* gA = A  + (size_t)(m0 + lrow) * D + lcol;
  const u16* gB = Wt + (size_t)(n0 + lrow) * D + lcol;

  // prologue: fill buffer 0 with the k=0 tiles
  cp16(&sA[0][lrow * LDT + lcol],     gA);
  cp16(&sA[0][lrow * LDT + lcol + 8], gA + 8);
  cp16(&sB[0][lrow * LDT + lcol],     gB);
  cp16(&sB[0][lrow * LDT + lcol + 8], gB + 8);
  wait_async();
  __syncthreads();

  int buf = 0;
  for (int k0 = 0; k0 < D; k0 += 32) {
    // kick off next tile into the other buffer (async; overlaps the WMMAs)
    const int nk = k0 + 32;
    if (nk < D) {
      cp16(&sA[buf ^ 1][lrow * LDT + lcol],     gA + nk);
      cp16(&sA[buf ^ 1][lrow * LDT + lcol + 8], gA + nk + 8);
      cp16(&sB[buf ^ 1][lrow * LDT + lcol],     gB + nk);
      cp16(&sB[buf ^ 1][lrow * LDT + lcol + 8], gB + nk + 8);
    }

    v16bf af[4], bfr[2];
#pragma unroll
    for (int rt = 0; rt < 4; ++rt) {
      int m = wr * 64 + rt * 16 + mlane;
      af[rt] = frag16(&sA[buf][m * LDT + khA], &sA[buf][m * LDT + 16 + khA]);
    }
#pragma unroll
    for (int ct = 0; ct < 2; ++ct) {
      int n = wc * 32 + ct * 16 + mlane;
      bfr[ct] = frag16(&sB[buf][n * LDT + khA], &sB[buf][n * LDT + 16 + khA]);
    }
#pragma unroll
    for (int rt = 0; rt < 4; ++rt)
#pragma unroll
      for (int ct = 0; ct < 2; ++ct)
        acc[rt][ct] = wmma_bf16(af[rt], bfr[ct], acc[rt][ct]);

    wait_async();      // this wave's async fills of buf^1 have landed
    __syncthreads();   // everyone done reading buf, everyone's fills visible
    buf ^= 1;
  }

  // epilogue: C layout -> VGPR r holds row (r + 8*hsel), col = mlane
  const int radd = hsel * 8;
#pragma unroll
  for (int rt = 0; rt < 4; ++rt) {
#pragma unroll
    for (int ct = 0; ct < 2; ++ct) {
      int gm = m0 + wr * 64 + rt * 16 + radd;
      int gn = n0 + wc * 32 + ct * 16 + mlane;
#pragma unroll
      for (int r = 0; r < 8; ++r) {
        float v = acc[rt][ct][r];
        size_t idx = (size_t)(gm + r) * D + gn;
        if (EPI == 0) outF[idx] = v;
        else          outB[idx] = f2bf(v);
      }
    }
  }
}

// ---------------- causal flash attention (bf16 WMMA, f32 softmax) ----------------
// Block: 128 threads (4 waves). Block handles one (b,h) and 64 query rows.
__global__ __launch_bounds__(128) void k_attn(
    const u16* __restrict__ Q, const u16* __restrict__ K, const u16* __restrict__ V,
    const float* __restrict__ bias,  // [NH][S][S]
    u16* __restrict__ O) {
  __shared__ __align__(16) u16 sVT[DH * 32];      // V^T tile: [dh][kv], 64x32
  __shared__ __align__(16) u16 sP[4][16 * 32];    // per-wave P tile: [qrow][kv]

  const int tid  = threadIdx.x;
  const int lane = tid & 31;
  const int wid  = tid >> 5;         // 0..3
  const int bh   = blockIdx.x;       // 0..31
  const int b    = bh >> 4;
  const int h    = bh & 15;
  const int q0   = blockIdx.y * 64;
  const int qb   = q0 + wid * 16;    // this wave's 16 query rows

  const int mlane = lane & 15;
  const int hsel  = lane >> 4;
  const int khA   = hsel * 8;
  const int kbB   = hsel * 16;

  // Q fragments (A operand), resident across the whole KV loop
  const u16* qr = Q + (size_t)(b * S + qb + mlane) * D + h * DH;
  v16bf qf[2];
  qf[0] = frag16(qr + khA,      qr + 16 + khA);
  qf[1] = frag16(qr + 32 + khA, qr + 48 + khA);

  v8f acc[4];
#pragma unroll
  for (int i = 0; i < 4; ++i) acc[i] = (v8f){0,0,0,0,0,0,0,0};
  float mrow[8], lrow[8];
#pragma unroll
  for (int r = 0; r < 8; ++r) { mrow[r] = -3.0e38f; lrow[r] = 0.0f; }

  const int kvend = q0 + 64;
  for (int kv0 = 0; kv0 < kvend; kv0 += 32) {
    __syncthreads();
    { // stage V^T: 32 kv rows x 64 dh, transposed into LDS
      int row = tid >> 2;            // 0..31
      int c0  = (tid & 3) * 16;
      const u16* vg = V + (size_t)(b * S + kv0 + row) * D + h * DH + c0;
#pragma unroll
      for (int j = 0; j < 16; ++j) sVT[(c0 + j) * 32 + row] = vg[j];
    }
    __syncthreads();

    // scores S = Q * K^T : 2 col tiles of 16 kv, K-dim = DH in two 32-steps
    v8f s[2];
#pragma unroll
    for (int ct = 0; ct < 2; ++ct) {
      const u16* kr = K + (size_t)(b * S + kv0 + ct * 16 + mlane) * D + h * DH;
      v16bf b0 = frag16(kr + khA,      kr + 16 + khA);
      v16bf b1 = frag16(kr + 32 + khA, kr + 48 + khA);
      v8f z = (v8f){0,0,0,0,0,0,0,0};
      z = wmma_bf16(qf[0], b0, z);
      z = wmma_bf16(qf[1], b1, z);
      s[ct] = z;
    }

    // bias + causal mask + online softmax (rows live in 16-lane groups)
#pragma unroll
    for (int r = 0; r < 8; ++r) {
      int grow = qb + hsel * 8 + r;
#pragma unroll
      for (int ct = 0; ct < 2; ++ct) {
        int gcol = kv0 + ct * 16 + mlane;
        float v = s[ct][r] * 0.125f +
                  bias[((size_t)h * S + grow) * S + gcol];
        if (gcol > grow) v = -3.0e38f;
        s[ct][r] = v;
      }
      float t = fmaxf(s[0][r], s[1][r]);
      t = fmaxf(t, __shfl_xor(t, 1, 32));
      t = fmaxf(t, __shfl_xor(t, 2, 32));
      t = fmaxf(t, __shfl_xor(t, 4, 32));
      t = fmaxf(t, __shfl_xor(t, 8, 32));
      float nm = fmaxf(mrow[r], t);
      s[0][r] = __expf(s[0][r] - nm);
      s[1][r] = __expf(s[1][r] - nm);
      float rs = s[0][r] + s[1][r];
      rs += __shfl_xor(rs, 1, 32);
      rs += __shfl_xor(rs, 2, 32);
      rs += __shfl_xor(rs, 4, 32);
      rs += __shfl_xor(rs, 8, 32);
      float corr = __expf(mrow[r] - nm);
      lrow[r] = lrow[r] * corr + rs;
      mrow[r] = nm;
#pragma unroll
      for (int cd = 0; cd < 4; ++cd) acc[cd][r] = acc[cd][r] * corr;
      // P (C-layout) -> LDS bf16 tile [16][32] for A-layout reload
      sP[wid][(hsel * 8 + r) * 32 + mlane]      = f2bf(s[0][r]);
      sP[wid][(hsel * 8 + r) * 32 + 16 + mlane] = f2bf(s[1][r]);
    }
    __syncthreads();  // also orders per-wave P store -> load

    // O += P(16x32) * V(32x64)
    v16bf pf = frag16(&sP[wid][mlane * 32 + khA], &sP[wid][mlane * 32 + 16 + khA]);
#pragma unroll
    for (int cd = 0; cd < 4; ++cd) {
      int n = cd * 16 + mlane;
      v16bf vf = frag16(&sVT[n * 32 + khA], &sVT[n * 32 + 16 + khA]);
      acc[cd] = wmma_bf16(pf, vf, acc[cd]);
    }
  }

  // normalize + write bf16 attention output [b][s][h*DH + dh]
#pragma unroll
  for (int r = 0; r < 8; ++r) {
    int grow = qb + hsel * 8 + r;
    float il = 1.0f / lrow[r];
#pragma unroll
    for (int cd = 0; cd < 4; ++cd) {
      int d = h * DH + cd * 16 + mlane;
      O[(size_t)(b * S + grow) * D + d] = f2bf(acc[cd][r] * il);
    }
  }
}

// ---------------- launch ----------------
extern "C" void kernel_launch(void* const* d_in, const int* in_sizes, int n_in,
                              void* d_out, int out_size, void* d_ws, size_t ws_size,
                              hipStream_t stream) {
  (void)in_sizes; (void)n_in; (void)out_size; (void)ws_size;
  const float* x     = (const float*)d_in[0];
  const float* basis = (const float*)d_in[1];
  const float* s_q   = (const float*)d_in[2];
  const float* s_k   = (const float*)d_in[3];
  const float* s_v   = (const float*)d_in[4];
  const float* s_o   = (const float*)d_in[5];
  const float* rbias = (const float*)d_in[6];

  char* w = (char*)d_ws;
  size_t off = 0;
  auto take = [&](size_t bytes) -> void* {
    void* p = w + off;
    off += (bytes + 255) & ~(size_t)255;
    return p;
  };
  u16*   xb = (u16*)take((size_t)M * D * 2);   // bf16 x
  u16*   bb = (u16*)take((size_t)D * D * 2);   // bf16 basis
  u16*   bt = (u16*)take((size_t)D * D * 2);   // bf16 basis^T
  float* yf = (float*)take((size_t)M * D * 4); // f32 intermediate
  u16*   tb = (u16*)take((size_t)M * D * 2);   // scaled bf16 temp
  u16*   qm = (u16*)take((size_t)M * D * 2);
  u16*   km = (u16*)take((size_t)M * D * 2);
  u16*   vm = (u16*)take((size_t)M * D * 2);
  u16*   am = (u16*)take((size_t)M * D * 2);   // attention output bf16

  const dim3 gG(D / 128, M / 128), tG(256);
  const int nMD = M * D, nDD = D * D;

  k_cvt      <<<nMD / 256, 256, 0, stream>>>(x, xb, nMD);
  k_cvt      <<<nDD / 256, 256, 0, stream>>>(basis, bb, nDD);
  k_transpose<<<nDD / 256, 256, 0, stream>>>(basis, bt);

  // y = x @ basis   (Wt = basis^T)
  k_gemm_bf16<0><<<gG, tG, 0, stream>>>(xb, bt, yf, nullptr);

  // q/k/v = (y * scale) @ basis^T   (Wt = basis)
  k_scale<<<nMD / 256, 256, 0, stream>>>(yf, s_q, tb, nMD);
  k_gemm_bf16<1><<<gG, tG, 0, stream>>>(tb, bb, nullptr, qm);
  k_scale<<<nMD / 256, 256, 0, stream>>>(yf, s_k, tb, nMD);
  k_gemm_bf16<1><<<gG, tG, 0, stream>>>(tb, bb, nullptr, km);
  k_scale<<<nMD / 256, 256, 0, stream>>>(yf, s_v, tb, nMD);
  k_gemm_bf16<1><<<gG, tG, 0, stream>>>(tb, bb, nullptr, vm);

  // causal attention with route bias
  k_attn<<<dim3(BS * NH, S / 64), 128, 0, stream>>>(qm, km, vm, rbias, am);

  // out = ((attn @ basis) * scale_o) @ basis^T
  k_gemm_bf16<0><<<gG, tG, 0, stream>>>(am, bt, yf, nullptr);
  k_scale<<<nMD / 256, 256, 0, stream>>>(yf, s_o, tb, nMD);
  k_gemm_bf16<0><<<gG, tG, 0, stream>>>(tb, bb, (float*)d_out, nullptr);
}
